// RandomizedPruningMasks_16174846836835
// MI455X (gfx1250) — matrix operand order repaired and checked by reading, compiled, not run on hardware
//
#include <hip/hip_runtime.h>

typedef __bf16 bf16;
typedef bf16  v16bf __attribute__((ext_vector_type(16)));
typedef bf16  v8bf  __attribute__((ext_vector_type(8)));
typedef float v8f   __attribute__((ext_vector_type(8)));

#define M_DIM 256
#define N_DIM 4096
#define K_DIM 4096
#define W_ELEMS (N_DIM * K_DIM)   // 16,777,216
#define X_ELEMS (M_DIM * K_DIM)   // 1,048,576
#define VALUE_SCALE 0.1f

// ---- split fp32 -> bf16 hi/lo (hi = rne(f), lo = rne(f - hi)), also clear claim ----
__global__ void cvt_split_w(const float* __restrict__ src, bf16* __restrict__ hi,
                            bf16* __restrict__ lo, int* __restrict__ claim, int n) {
  int i = blockIdx.x * blockDim.x + threadIdx.x;
  if (i < n) {
    float f = src[i];
    bf16 h = (bf16)f;
    hi[i] = h;
    lo[i] = (bf16)(f - (float)h);
    claim[i] = 0;
  }
}

__global__ void cvt_split_x(const float* __restrict__ src, bf16* __restrict__ hi,
                            bf16* __restrict__ lo, int n) {
  int i = blockIdx.x * blockDim.x + threadIdx.x;
  if (i < n) {
    float f = src[i];
    bf16 h = (bf16)f;
    hi[i] = h;
    lo[i] = (bf16)(f - (float)h);
  }
}

// ---- deterministic scatter: highest flip-list position wins ----
__global__ void scatter_claim(const int* __restrict__ idx, int* __restrict__ claim, int n) {
  int i = blockIdx.x * blockDim.x + threadIdx.x;
  if (i < n) atomicMax(claim + idx[i], i + 1);
}

__global__ void scatter_write(const int* __restrict__ idx, const float* __restrict__ vals,
                              const int* __restrict__ claim, bf16* __restrict__ whi,
                              bf16* __restrict__ wlo, int n) {
  int i = blockIdx.x * blockDim.x + threadIdx.x;
  if (i < n) {
    int j = idx[i];
    if (claim[j] == i + 1) {
      float v = vals[i] * VALUE_SCALE;
      bf16 h = (bf16)v;
      whi[j] = h;
      wlo[j] = (bf16)(v - (float)h);
    }
  }
}

// ---- bf16x3 WMMA GEMM: out[m,n] = sum_k x[m,k] * W[n,k] ----
// grid: (N_DIM/64, M_DIM/128), block: 256 (8 waves). Wave w -> M-tile (by*8+w),
// all 8 waves in a block share the same 64-column N strip (B reuse in L0/L2).
__global__ __launch_bounds__(256) void gemm_bf16x3(const bf16* __restrict__ x_hi,
                                                   const bf16* __restrict__ x_lo,
                                                   const bf16* __restrict__ w_hi,
                                                   const bf16* __restrict__ w_lo,
                                                   float* __restrict__ out) {
  const int lane   = threadIdx.x & 31;
  const int wave   = threadIdx.x >> 5;
  const int m_tile = blockIdx.y * 8 + wave;
  const int n_base = blockIdx.x * 64;

  // A fragment (contiguous k order, same as B): lane m = lane&15; lanes 0-15 cover K 0..15,
  // lanes 16-31 cover K 16..31  -> two 16B contiguous loads.
  const int    mrow = m_tile * 16 + (lane & 15);
  const int    akb  = (lane & 16) ? 16 : 0;
  const bf16* xh = x_hi + (size_t)mrow * K_DIM + akb;
  const bf16* xl = x_lo + (size_t)mrow * K_DIM + akb;

  // B fragment (32x16): lane n = lane&15; lanes 0-15 hold K 0..15, lanes 16-31
  // hold K 16..31. B[k,n] = W[n,k] row-major -> one contiguous 32B load.
  const int bkb = (lane & 16) ? 16 : 0;
  const bf16* wh[4];
  const bf16* wl[4];
#pragma unroll
  for (int t = 0; t < 4; ++t) {
    size_t off = (size_t)(n_base + t * 16 + (lane & 15)) * K_DIM + bkb;
    wh[t] = w_hi + off;
    wl[t] = w_lo + off;
  }

  v8f acc[4];
#pragma unroll
  for (int t = 0; t < 4; ++t)
    acc[t] = (v8f){0.f, 0.f, 0.f, 0.f, 0.f, 0.f, 0.f, 0.f};

  for (int k = 0; k < K_DIM; k += 32) {
    v8bf ah0 = *(const v8bf*)(xh + k);
    v8bf ah1 = *(const v8bf*)(xh + k + 8);
    v8bf al0 = *(const v8bf*)(xl + k);
    v8bf al1 = *(const v8bf*)(xl + k + 8);
    v16bf ah = __builtin_shufflevector(ah0, ah1, 0,1,2,3,4,5,6,7,8,9,10,11,12,13,14,15);
    v16bf al = __builtin_shufflevector(al0, al1, 0,1,2,3,4,5,6,7,8,9,10,11,12,13,14,15);
#pragma unroll
    for (int t = 0; t < 4; ++t) {
      v16bf bh = *(const v16bf*)(wh[t] + k);
      v16bf bl = *(const v16bf*)(wl[t] + k);
      // fp32-fidelity product: hi*hi + hi*lo + lo*hi (lo*lo negligible)
      acc[t] = __builtin_amdgcn_wmma_f32_16x16x32_bf16(false, ah, false, bh,
                                                       (short)0, acc[t], false, false);
      acc[t] = __builtin_amdgcn_wmma_f32_16x16x32_bf16(false, ah, false, bl,
                                                       (short)0, acc[t], false, false);
      acc[t] = __builtin_amdgcn_wmma_f32_16x16x32_bf16(false, al, false, bh,
                                                       (short)0, acc[t], false, false);
    }
  }

  // C/D layout: VGPR r, lanes 0-15 -> row r, col lane; lanes 16-31 -> row 8+r.
  const int orow = m_tile * 16 + ((lane & 16) ? 8 : 0);
#pragma unroll
  for (int t = 0; t < 4; ++t) {
    const int ocol = n_base + t * 16 + (lane & 15);
#pragma unroll
    for (int r = 0; r < 8; ++r)
      out[(size_t)(orow + r) * N_DIM + ocol] = acc[t][r];
  }
}

extern "C" void kernel_launch(void* const* d_in, const int* in_sizes, int n_in,
                              void* d_out, int out_size, void* d_ws, size_t ws_size,
                              hipStream_t stream) {
  const float* x    = (const float*)d_in[0];
  const float* w    = (const float*)d_in[1];
  const int*   flip = (const int*)d_in[2];
  const float* vals = (const float*)d_in[3];
  float*       out  = (float*)d_out;
  const int nflip = in_sizes[2];

  // workspace layout (bytes)
  char* ws = (char*)d_ws;
  bf16* w_hi  = (bf16*)(ws);                                 // 32 MiB
  bf16* w_lo  = (bf16*)(ws + (size_t)W_ELEMS * 2);           // 32 MiB
  bf16* x_hi  = (bf16*)(ws + (size_t)W_ELEMS * 4);           // 2 MiB
  bf16* x_lo  = (bf16*)(ws + (size_t)W_ELEMS * 4 + (size_t)X_ELEMS * 2);
  int*  claim = (int*) (ws + (size_t)W_ELEMS * 4 + (size_t)X_ELEMS * 4); // 64 MiB

  cvt_split_w<<<W_ELEMS / 256, 256, 0, stream>>>(w, w_hi, w_lo, claim, W_ELEMS);
  cvt_split_x<<<X_ELEMS / 256, 256, 0, stream>>>(x, x_hi, x_lo, X_ELEMS);
  scatter_claim<<<(nflip + 255) / 256, 256, 0, stream>>>(flip, claim, nflip);
  scatter_write<<<(nflip + 255) / 256, 256, 0, stream>>>(flip, vals, claim, w_hi, w_lo, nflip);

  dim3 grid(N_DIM / 64, M_DIM / 128);   // 64 x 2 blocks
  gemm_bf16x3<<<grid, 256, 0, stream>>>(x_hi, x_lo, w_hi, w_lo, out);
}
